// GRUPolicy_71743133712822
// MI455X (gfx1250) — matrix operand order repaired and checked by reading, not compiled, not run on hardware
//
#include <hip/hip_runtime.h>
#include <hip/hip_bf16.h>
#include <math.h>

// ---------------- problem constants ----------------
#define BB   64
#define TT   512
#define OBS  128
#define FEAT 512
#define HH   256
#define GG   768           // 3*H
#define BT   32768         // B*T
#define NACT 10

typedef __attribute__((ext_vector_type(16))) __bf16 v16bf;
typedef __attribute__((ext_vector_type(8)))  float  v8f;

union ABu { v16bf v; unsigned u[8]; };

// round-to-nearest-even f32 -> bf16 bits
__device__ __forceinline__ unsigned short f2bf(float f) {
    unsigned u = __float_as_uint(f);
    unsigned r = (u + 0x7FFFu + ((u >> 16) & 1u)) >> 16;
    return (unsigned short)r;
}
__device__ __forceinline__ float bf2f(unsigned short b) {
    return __uint_as_float(((unsigned)b) << 16);
}
__device__ __forceinline__ float sigmoidf_(float x) {
    return 1.0f / (1.0f + expf(-x));
}

// Load 16x32 bf16 A tile (row-major source, row stride ld elements).
// Lane L: M = L&15, half = L>>4; VGPR v holds K = (v/4)*16 + half*8 + (v%4)*2 (+0,+1)
__device__ __forceinline__ v16bf load_a_bf16(const unsigned short* base, int ld) {
    int lane = threadIdx.x & 31;
    int m    = lane & 15;
    int half = lane >> 4;
    ABu t;
#pragma unroll
    for (int v = 0; v < 8; ++v) {
        int k = ((v >> 2) << 4) + (half << 3) + ((v & 3) << 1);
        t.u[v] = *(const unsigned*)(base + (size_t)m * ld + k);
    }
    return t.v;
}

// Load 32x16 bf16 B tile for D = A x B where B[k][n] = W[n][k], W row-major (N x K).
// base points at W row (ntile*16), element k0. Lane L: N = L&15, K = (v/4)*16 + (L>>4)*8 + (v%4)*2 (+0,+1)
__device__ __forceinline__ v16bf load_b_bf16(const unsigned short* base, int ld) {
    int lane = threadIdx.x & 31;
    int n    = lane & 15;
    int half = lane >> 4;
    ABu t;
#pragma unroll
    for (int v = 0; v < 8; ++v) {
        int k = ((v >> 2) << 4) + (half << 3) + ((v & 3) << 1);
        t.u[v] = *(const unsigned*)(base + (size_t)n * ld + k);
    }
    return t.v;
}

// ---------------- kernel 0: f32 -> bf16 convert ----------------
__global__ __launch_bounds__(256) void cvt_bf16_kernel(const float* __restrict__ src,
                                                       unsigned short* __restrict__ dst, int n) {
    int i = blockIdx.x * 256 + threadIdx.x;
    if (i < n) dst[i] = f2bf(src[i]);
}

// ---------------- kernel 1: feat = relu(obs @ W1^T + b1), bf16 out ----------------
// M=BT, N=FEAT, K=OBS. 2x2 macro-tile (32x32) per wave: 4 tile loads -> 4 WMMAs per K step.
__global__ __launch_bounds__(256) void feat_kernel(const unsigned short* __restrict__ obs_bf,
                                                   const unsigned short* __restrict__ W1_bf,
                                                   const float* __restrict__ b1,
                                                   unsigned short* __restrict__ feat_bf) {
    int wave = threadIdx.x >> 5;
    int job  = blockIdx.x * 8 + wave;          // (BT/32)*(FEAT/32) = 1024*16 jobs
    int mi2  = job >> 4;                        // macro row  (32 rows)
    int ni2  = job & 15;                        // macro col  (32 cols)
    const unsigned short* arow0 = obs_bf + (size_t)(mi2 * 32) * OBS;
    const unsigned short* arow1 = arow0 + (size_t)16 * OBS;
    const unsigned short* brow0 = W1_bf + (size_t)(ni2 * 32) * OBS;
    const unsigned short* brow1 = brow0 + (size_t)16 * OBS;
    v8f c00 = {}, c01 = {}, c10 = {}, c11 = {};
#pragma unroll
    for (int kb = 0; kb < 4; ++kb) {
        int k0 = kb * 32;
        v16bf a0 = load_a_bf16(arow0 + k0, OBS);
        v16bf a1 = load_a_bf16(arow1 + k0, OBS);
        v16bf b0 = load_b_bf16(brow0 + k0, OBS);
        v16bf b1v = load_b_bf16(brow1 + k0, OBS);
        c00 = __builtin_amdgcn_wmma_f32_16x16x32_bf16(false, a0, false, b0,  (short)0, c00, false, false);
        c01 = __builtin_amdgcn_wmma_f32_16x16x32_bf16(false, a0, false, b1v, (short)0, c01, false, false);
        c10 = __builtin_amdgcn_wmma_f32_16x16x32_bf16(false, a1, false, b0,  (short)0, c10, false, false);
        c11 = __builtin_amdgcn_wmma_f32_16x16x32_bf16(false, a1, false, b1v, (short)0, c11, false, false);
    }
    int lane = threadIdx.x & 31;
    int half = lane >> 4;
    int nl   = lane & 15;
    v8f accs[4] = {c00, c01, c10, c11};
#pragma unroll
    for (int q = 0; q < 4; ++q) {
        int m0 = mi2 * 32 + (q >> 1) * 16;
        int n  = ni2 * 32 + (q & 1) * 16 + nl;
        float bias = b1[n];
#pragma unroll
        for (int cc = 0; cc < 8; ++cc) {
            int m = m0 + (half << 3) + cc;
            float x = accs[q][cc] + bias;
            x = x > 0.0f ? x : 0.0f;
            feat_bf[(size_t)m * FEAT + n] = f2bf(x);
        }
    }
}

// ---------------- kernel 2: gx[t][b][g] = feat @ W_ih^T + b_ih (f32 out) ----------------
// M=BT, N=GG, K=FEAT. 2x2 macro-tile per wave. Row m = b*T + t; stored at ((t*B)+b)*G + g.
__global__ __launch_bounds__(256) void gx_kernel(const unsigned short* __restrict__ feat_bf,
                                                 const unsigned short* __restrict__ Wih_bf,
                                                 const float* __restrict__ b_ih,
                                                 float* __restrict__ gx) {
    int wave = threadIdx.x >> 5;
    int job  = blockIdx.x * 8 + wave;          // (BT/32)*(GG/32) = 1024*24 jobs
    int mi2  = job / 24;
    int ni2  = job % 24;
    const unsigned short* arow0 = feat_bf + (size_t)(mi2 * 32) * FEAT;
    const unsigned short* arow1 = arow0 + (size_t)16 * FEAT;
    const unsigned short* brow0 = Wih_bf + (size_t)(ni2 * 32) * FEAT;
    const unsigned short* brow1 = brow0 + (size_t)16 * FEAT;
    v8f c00 = {}, c01 = {}, c10 = {}, c11 = {};
#pragma unroll 4
    for (int kb = 0; kb < 16; ++kb) {
        int k0 = kb * 32;
        v16bf a0 = load_a_bf16(arow0 + k0, FEAT);
        v16bf a1 = load_a_bf16(arow1 + k0, FEAT);
        v16bf b0 = load_b_bf16(brow0 + k0, FEAT);
        v16bf b1v = load_b_bf16(brow1 + k0, FEAT);
        c00 = __builtin_amdgcn_wmma_f32_16x16x32_bf16(false, a0, false, b0,  (short)0, c00, false, false);
        c01 = __builtin_amdgcn_wmma_f32_16x16x32_bf16(false, a0, false, b1v, (short)0, c01, false, false);
        c10 = __builtin_amdgcn_wmma_f32_16x16x32_bf16(false, a1, false, b0,  (short)0, c10, false, false);
        c11 = __builtin_amdgcn_wmma_f32_16x16x32_bf16(false, a1, false, b1v, (short)0, c11, false, false);
    }
    int lane = threadIdx.x & 31;
    int half = lane >> 4;
    int nl   = lane & 15;
    v8f accs[4] = {c00, c01, c10, c11};
#pragma unroll
    for (int q = 0; q < 4; ++q) {
        int m0 = mi2 * 32 + (q >> 1) * 16;
        int n  = ni2 * 32 + (q & 1) * 16 + nl;
        float bias = b_ih[n];
#pragma unroll
        for (int cc = 0; cc < 8; ++cc) {
            int m = m0 + (half << 3) + cc;
            int b = m >> 9;          // m / T
            int t = m & 511;         // m % T
            gx[((size_t)t * BB + b) * GG + n] = accs[q][cc] + bias;
        }
    }
}

// ---------------- kernel 3: sequential GRU scan (single 32-wave workgroup) ----------------
// Dynamic LDS: 64KB f32 h state + 32KB bf16 shadow (WMMA A operand source).
__global__ __launch_bounds__(1024) void gru_kernel(const float* __restrict__ gx,
                                                   const int* __restrict__ masks,
                                                   const float* __restrict__ h0,
                                                   const unsigned short* __restrict__ Whh_bf,
                                                   const float* __restrict__ b_hh,
                                                   float* __restrict__ hs) {
    extern __shared__ char smem[];
    float*          h   = (float*)smem;                       // 64 KB
    unsigned short* hbf = (unsigned short*)(smem + BB * HH * 4); // 32 KB

    const int tid  = threadIdx.x;
    const int lane = tid & 31;
    const int wave = tid >> 5;
    const int half = lane >> 4;
    const int mi   = wave >> 3;            // batch tile 0..3 (shared by both gate tiles)
    const int ji0  = (wave & 7) * 2;       // first of two consecutive column tiles

    // init state + bf16 shadow; pre-warm W_hh (384KB, L2-resident afterwards)
    for (int i = tid; i < BB * HH; i += 1024) {
        float v = h0[i];
        h[i]   = v;
        hbf[i] = f2bf(v);
    }
    {
        const char* w = (const char*)Whh_bf;
        for (int i = tid; i < (GG * HH * 2) / 64; i += 1024)
            __builtin_prefetch(w + i * 64, 0, 1);
    }
    __syncthreads();

    for (int t = 0; t < TT; ++t) {
        // Phase A: h *= mask_t (mask is exactly 0/1) ; mask_0 == 1
        if (t > 0) {
            for (int i = tid; i < BB * HH; i += 1024) {
                int b = i >> 8;
                if (masks[b * (TT - 1) + (t - 1)] == 0) { h[i] = 0.0f; hbf[i] = 0; }
            }
        }
        // prefetch next step's gx slab (192KB) while this step computes
        if (t + 1 < TT) {
            const char* p = (const char*)(gx + (size_t)(t + 1) * BB * GG);
            for (int i = tid; i < (BB * GG * 4) / 64; i += 1024)
                __builtin_prefetch(p + i * 64, 0, 1);
        }
        __syncthreads();

        // Phase B: gh = h @ W_hh^T.  One A tile per K-step feeds 6 WMMAs (r/z/n x 2 cols).
        v8f acc[6];
#pragma unroll
        for (int i = 0; i < 6; ++i) { v8f z = {}; acc[i] = z; }
#pragma unroll
        for (int kb = 0; kb < 8; ++kb) {
            int k0 = kb * 32;
            v16bf a = load_a_bf16(&hbf[(mi * 16) * HH + k0], HH);
#pragma unroll
            for (int g = 0; g < 2; ++g) {
                int ji = ji0 + g;
                v16bf br = load_b_bf16(Whh_bf + (size_t)((ji      ) * 16) * HH + k0, HH);
                v16bf bz = load_b_bf16(Whh_bf + (size_t)((ji + 16) * 16) * HH + k0, HH);
                v16bf bn = load_b_bf16(Whh_bf + (size_t)((ji + 32) * 16) * HH + k0, HH);
                acc[g * 3 + 0] = __builtin_amdgcn_wmma_f32_16x16x32_bf16(false, a, false, br, (short)0, acc[g * 3 + 0], false, false);
                acc[g * 3 + 1] = __builtin_amdgcn_wmma_f32_16x16x32_bf16(false, a, false, bz, (short)0, acc[g * 3 + 1], false, false);
                acc[g * 3 + 2] = __builtin_amdgcn_wmma_f32_16x16x32_bf16(false, a, false, bn, (short)0, acc[g * 3 + 2], false, false);
            }
        }
        __syncthreads();   // all reads of h/hbf done before gate updates write them

        // Phase C: gates + state update + stream hs
#pragma unroll
        for (int g = 0; g < 2; ++g) {
            int ji = ji0 + g;
            int j  = ji * 16 + (lane & 15);
            float bhr = b_hh[j];
            float bhz = b_hh[j + HH];
            float bhn = b_hh[j + 2 * HH];
#pragma unroll
            for (int cc = 0; cc < 8; ++cc) {
                int b = mi * 16 + (half << 3) + cc;
                const float* gxp = gx + ((size_t)t * BB + b) * GG + j;
                float r = sigmoidf_(gxp[0]      + acc[g * 3 + 0][cc] + bhr);
                float z = sigmoidf_(gxp[HH]     + acc[g * 3 + 1][cc] + bhz);
                float n = tanhf   (gxp[2 * HH] + r * (acc[g * 3 + 2][cc] + bhn));
                float hold = h[b * HH + j];     // masked h (f32 carry)
                float hn = (1.0f - z) * n + z * hold;
                h[b * HH + j]   = hn;
                hbf[b * HH + j] = f2bf(hn);
                hs[((size_t)b * TT + t) * HH + j] = hn;
            }
        }
        __syncthreads();
    }
}

// ---------------- kernel 4: policy/value heads (one wave per row) ----------------
__global__ __launch_bounds__(256) void heads_kernel(const float* __restrict__ hs,
                                                    const float* __restrict__ Wa,
                                                    const float* __restrict__ ba,
                                                    const float* __restrict__ Wc,
                                                    const float* __restrict__ bc,
                                                    const int* __restrict__ action,
                                                    float* __restrict__ out) {
    int row  = blockIdx.x * 8 + (threadIdx.x >> 5);
    int lane = threadIdx.x & 31;
    float p[NACT + 1];
#pragma unroll
    for (int a = 0; a <= NACT; ++a) p[a] = 0.0f;
    const float* hrow = hs + (size_t)row * HH;
    for (int e = lane; e < HH; e += 32) {
        float hv = hrow[e];
#pragma unroll
        for (int a = 0; a < NACT; ++a) p[a] += Wa[a * HH + e] * hv;
        p[NACT] += Wc[e] * hv;
    }
#pragma unroll
    for (int off = 16; off > 0; off >>= 1) {
#pragma unroll
        for (int a = 0; a <= NACT; ++a) p[a] += __shfl_xor(p[a], off, 32);
    }
    if (lane == 0) {
        float logits[NACT];
        float maxv = -3.4e38f;
#pragma unroll
        for (int a = 0; a < NACT; ++a) { logits[a] = p[a] + ba[a]; maxv = fmaxf(maxv, logits[a]); }
        float s = 0.0f;
#pragma unroll
        for (int a = 0; a < NACT; ++a) s += expf(logits[a] - maxv);
        float lse = maxv + logf(s);
        float ent = 0.0f;
#pragma unroll
        for (int a = 0; a < NACT; ++a) { float lp = logits[a] - lse; ent -= expf(lp) * lp; }
        int act = action[row];
        out[row]            = p[NACT] + bc[0];          // value
        out[BT + row]       = logits[act] - lse;        // log_prob
        out[2 * BT + row]   = ent;                      // entropy
    }
}

// ---------------- host launch ----------------
extern "C" void kernel_launch(void* const* d_in, const int* in_sizes, int n_in,
                              void* d_out, int out_size, void* d_ws, size_t ws_size,
                              hipStream_t stream) {
    const float* obs   = (const float*)d_in[0];
    const int*   act   = (const int*)d_in[1];
    const int*   masks = (const int*)d_in[2];
    const float* oldh  = (const float*)d_in[3];
    const float* W1    = (const float*)d_in[4];
    const float* b1    = (const float*)d_in[5];
    const float* W_ih  = (const float*)d_in[6];
    const float* W_hh  = (const float*)d_in[7];
    const float* b_ih  = (const float*)d_in[8];
    const float* b_hh  = (const float*)d_in[9];
    const float* Wa    = (const float*)d_in[10];
    const float* ba    = (const float*)d_in[11];
    const float* Wc    = (const float*)d_in[12];
    const float* bc    = (const float*)d_in[13];
    float* out = (float*)d_out;

    char* ws = (char*)d_ws;
    unsigned short* obs_bf  = (unsigned short*)(ws);                         //  8 MB
    unsigned short* W1_bf   = (unsigned short*)(ws + 8388608);               // 128 KB
    unsigned short* Wih_bf  = (unsigned short*)(ws + 8519680);               // 768 KB
    unsigned short* Whh_bf  = (unsigned short*)(ws + 9306112);               // 384 KB
    unsigned short* feat_bf = (unsigned short*)(ws + 9699328);               // 32 MB
    float*          gx      = (float*)        (ws + 43253760);               // 96 MB
    float*          hs      = (float*)        (ws + 143917056);              // 32 MB

    // 0) conversions
    cvt_bf16_kernel<<<(BT * OBS + 255) / 256, 256, 0, stream>>>(obs,  obs_bf,  BT * OBS);
    cvt_bf16_kernel<<<(FEAT * OBS + 255) / 256, 256, 0, stream>>>(W1,  W1_bf,  FEAT * OBS);
    cvt_bf16_kernel<<<(GG * FEAT + 255) / 256, 256, 0, stream>>>(W_ih, Wih_bf, GG * FEAT);
    cvt_bf16_kernel<<<(GG * HH + 255) / 256, 256, 0, stream>>>(W_hh, Whh_bf, GG * HH);

    // 1) feat = relu(obs @ W1^T + b1)   (1024*16 2x2 jobs / 8 waves per block)
    feat_kernel<<<2048, 256, 0, stream>>>(obs_bf, W1_bf, b1, feat_bf);

    // 2) gx = feat @ W_ih^T + b_ih      (1024*24 2x2 jobs / 8 waves per block)
    gx_kernel<<<3072, 256, 0, stream>>>(feat_bf, Wih_bf, b_ih, gx);

    // 3) sequential GRU scan, one 32-wave workgroup, 96KB dynamic LDS
    gru_kernel<<<1, 1024, 96 * 1024, stream>>>(gx, masks, oldh, Whh_bf, b_hh, hs);

    // 4) heads
    heads_kernel<<<BT / 8, 256, 0, stream>>>(hs, Wa, ba, Wc, bc, act, out);
}
